// FullContextMemoryBank_87926570483963
// MI455X (gfx1250) — matrix operand order repaired and checked by reading, not compiled, not run on hardware
//
#include <hip/hip_runtime.h>
#include <math.h>

// ---------------------------------------------------------------------------
// FullContextMemoryBank for MI455X (gfx1250, wave32, WMMA).
// All 1x1-conv GEMMs run on v_wmma_f32_16x16x32_f16 (f16 in, f32 accumulate).
// LDS tiles are stored in WMMA fragment order so fragment loads are b128.
// ---------------------------------------------------------------------------

typedef _Float16 half_t;
typedef __attribute__((ext_vector_type(16))) _Float16 v16h;
typedef __attribute__((ext_vector_type(8)))  _Float16 h8;
typedef __attribute__((ext_vector_type(8)))  float    v8f;

#define B_      8
#define C_      640
#define H_      64
#define W_      64
#define HW_     4096
#define KD_     160
#define MR_     4
#define VR_     8
#define GROUPS_ 32
#define CPG_    20      // C_/GROUPS_

#define LDA_ 40         // halves per A row (32 data + 8 pad); 80B = 16B-multiple
#define LDB_ 40         // halves per B row (k-major, transposed)

__device__ __forceinline__ float gelu_f(float v) {
  return 0.5f * v * (1.0f + erff(v * 0.70710678118654752f));
}
__device__ __forceinline__ float sigmoid_f(float v) {
  return 1.0f / (1.0f + expf(-v));
}

// ---------------------------------------------------------------------------
// Wave-cooperative GEMM: block tile 64(M) x 64(N), K stepped by 32.
// 8 waves (wm 0..3, wn 0..1), each owns 16(M) x 32(N) as two 16x16 WMMAs.
// A = W[m, k] row-major (ld=ldw), B = X[k, p] row-major (ld=ldx).
//
// LDS A layout: row m holds its 32 k-values in natural k order (no permutation)
// so lane (m, half) reads 16 contiguous halves (2x b128).
// LDS B layout: k-major Bs[p][k]; lane (n, half) reads 16 contiguous halves.
// ---------------------------------------------------------------------------
__device__ __forceinline__ void gemm_acc(
    half_t* __restrict__ As, half_t* __restrict__ Bs,
    const float* __restrict__ W, int ldw, int kw0,
    const float* __restrict__ X, int ldx,
    int m0, int Mtot, int p0, int Ksz, v8f* acc)
{
  const int tid  = threadIdx.x;
  const int lane = tid & 31;
  const int wv   = tid >> 5;
  const int wm   = wv & 3;
  const int wn   = wv >> 2;

  // A staging: thread -> (row 0..63, source 8-block of k)
  const int arow  = tid >> 2;
  const int acb   = (tid & 3) * 8;                                   // 0,8,16,24
  const int adst  = acb;                                             // 0,8,16,24
  const bool aval = (m0 + arow) < Mtot;
  const int agm   = aval ? (m0 + arow) : (Mtot - 1);                 // clamped, always legal
  const float* Wp = W + (long)agm * ldw + kw0 + acb;
  half_t* AsW     = As + arow * LDA_ + adst;

  // B staging: thread -> (p 0..63, 8-block of k); p-coalesced global loads,
  // transposed (k-contiguous) b128 store.
  const int bp    = tid & 63;
  const int bkb   = (tid >> 6) * 8;                                  // 0,8,16,24
  const float* Xp = X + (long)bkb * ldx + p0 + bp;
  half_t* BsW     = Bs + bp * LDB_ + bkb;

  // Fragment read addresses (all 16B-aligned, contiguous 32B per fragment)
  const int ah = (lane >> 4) * 16;
  const half_t* AsR = As + (wm * 16 + (lane & 15)) * LDA_ + ah;
  const half_t* B0R = Bs + (wn * 32 + (lane & 15)) * LDB_ + ah;
  const half_t* B1R = B0R + 16 * LDB_;

  const int nsteps = Ksz >> 5;
  for (int kt = 0; kt < nsteps; ++kt) {
    // ---- stage A: 8 consecutive f32 (vectorizes to 2x global b128) ----
    {
      const float* s = Wp + (kt << 5);
      float f[8];
#pragma unroll
      for (int i = 0; i < 8; ++i) f[i] = s[i];
      h8 v;
#pragma unroll
      for (int i = 0; i < 8; ++i) v[i] = (half_t)(aval ? f[i] : 0.0f);
      *(h8*)AsW = v;
    }
    // ---- stage B: 8 k-strided dword loads (p-coalesced across lanes) ----
    {
      const float* s = Xp + (long)(kt << 5) * ldx;
      float f[8];
#pragma unroll
      for (int i = 0; i < 8; ++i) f[i] = s[(long)i * ldx];
      h8 v;
#pragma unroll
      for (int i = 0; i < 8; ++i) v[i] = (half_t)f[i];
      *(h8*)BsW = v;
    }
    __syncthreads();
    // prefetch next K tile into cache (gfx1250 global_prefetch_b8)
    if (kt + 1 < nsteps) {
      __builtin_prefetch(Wp + ((kt + 1) << 5), 0, 1);
      __builtin_prefetch(Xp + (long)((kt + 1) << 5) * ldx, 0, 1);
    }
    // ---- fragment loads: 6x ds_load_b128 per wave ----
    h8 a0 = *(const h8*)(AsR);
    h8 a1 = *(const h8*)(AsR + 8);
    h8 c0 = *(const h8*)(B0R);
    h8 c1 = *(const h8*)(B0R + 8);
    h8 d0 = *(const h8*)(B1R);
    h8 d1 = *(const h8*)(B1R + 8);
    v16h a, b0, b1;
#pragma unroll
    for (int j = 0; j < 8; ++j) {
      a[j]  = a0[j]; a[j + 8]  = a1[j];
      b0[j] = c0[j]; b0[j + 8] = c1[j];
      b1[j] = d0[j]; b1[j + 8] = d1[j];
    }
    acc[0] = __builtin_amdgcn_wmma_f32_16x16x32_f16(false, a, false, b0,
                                                    (short)0, acc[0], false, false);
    acc[1] = __builtin_amdgcn_wmma_f32_16x16x32_f16(false, a, false, b1,
                                                    (short)0, acc[1], false, false);
    __syncthreads();
  }
}

// C/D tile coordinates: VGPR r holds M = r (lanes 0-15) or r+8 (lanes 16-31),
// N = lane & 15.
#define TILE_MN(m_, n_, sub_, r_)                                   \
  const int m_ = (wv & 3) * 16 + (r_) + 8 * (lane >> 4);            \
  const int n_ = (wv >> 2) * 32 + (sub_) * 16 + (lane & 15);

// ---------------------------------------------------------------------------
__global__ void k_zero(float* p, int n) {
  int i = blockIdx.x * blockDim.x + threadIdx.x;
  if (i < n) p[i] = 0.0f;
}

// ---- query projection GEMM + GELU + spatial-mean accumulation -------------
__global__ void __launch_bounds__(256) k_qproj(
    const float* __restrict__ x, const float* __restrict__ qp_w,
    const float* __restrict__ qp_b, float* __restrict__ qsum)
{
  __shared__ __align__(16) half_t As[64 * LDA_];
  __shared__ __align__(16) half_t Bs[64 * LDB_];
  const int p0 = blockIdx.x * 64;
  const int m0 = blockIdx.y * 64;
  const int b  = blockIdx.z;
  v8f acc[2] = {};
  gemm_acc(As, Bs, qp_w, C_, 0, x + (long)b * C_ * HW_, HW_,
           m0, KD_, p0, C_, acc);
  const int lane = threadIdx.x & 31;
  const int wv   = threadIdx.x >> 5;
#pragma unroll
  for (int r = 0; r < 8; ++r) {
    const int m = m0 + (wv & 3) * 16 + r + 8 * (lane >> 4);
    const float bias = (m < KD_) ? qp_b[m] : 0.0f;
    float tot = gelu_f(acc[0][r] + bias) + gelu_f(acc[1][r] + bias);
    // reduce over the 16 N-lanes that share this m (wave32 halves)
    for (int mk = 1; mk < 16; mk <<= 1) tot += __shfl_xor(tot, mk, 32);
    if ((lane & 15) == 0 && m < KD_) atomicAdd(&qsum[b * KD_ + m], tot);
  }
}

// ---- cosine sims, top-2, softmax ------------------------------------------
__global__ void k_attn(const float* __restrict__ qsum,
                       const float* __restrict__ mem_keys,
                       float* __restrict__ attn, int* __restrict__ idx)
{
  const int b = threadIdx.x;
  if (b >= B_) return;
  const float inv = 1.0f / (float)HW_;
  float nq = 0.0f;
  for (int j = 0; j < KD_; ++j) { float q = qsum[b * KD_ + j] * inv; nq += q * q; }
  nq = fmaxf(sqrtf(nq), 1e-12f);
  float sims[MR_];
  for (int r = 0; r < MR_; ++r) {
    float nk = 0.0f, dot = 0.0f;
    for (int j = 0; j < KD_; ++j) {
      float kv = mem_keys[r * KD_ + j];
      float qv = qsum[b * KD_ + j] * inv;
      nk += kv * kv; dot += qv * kv;
    }
    sims[r] = dot / (nq * fmaxf(sqrtf(nk), 1e-12f));
  }
  int i0 = 0;
  for (int r = 1; r < MR_; ++r) if (sims[r] > sims[i0]) i0 = r;
  int i1 = (i0 == 0) ? 1 : 0;
  for (int r = 0; r < MR_; ++r) if (r != i0 && sims[r] > sims[i1]) i1 = r;
  const float v0 = sims[i0] * 10.0f, v1 = sims[i1] * 10.0f;   // /0.1 temperature
  const float e1 = expf(v1 - v0);                             // v0 is the max
  const float s  = 1.0f + e1;
  attn[b * 2 + 0] = 1.0f / s;
  attn[b * 2 + 1] = e1 / s;
  idx[b * 2 + 0] = i0;
  idx[b * 2 + 1] = i1;
}

// ---- decode conv1x1 on the 8x8 refs (conv commutes with bilinear) ---------
__global__ void __launch_bounds__(256) k_refproj(
    const float* __restrict__ mem_values, const float* __restrict__ vd_w1,
    const float* __restrict__ vd_b1, const int* __restrict__ idx,
    float* __restrict__ y)
{
  __shared__ __align__(16) half_t As[64 * LDA_];
  __shared__ __align__(16) half_t Bs[64 * LDB_];
  const int m0 = blockIdx.y * 64;
  const int bk = blockIdx.z;
  const float* X = mem_values + (long)idx[bk] * C_ * (VR_ * VR_);
  v8f acc[2] = {};
  gemm_acc(As, Bs, vd_w1, C_, 0, X, VR_ * VR_, m0, C_, 0, C_, acc);
  const int lane = threadIdx.x & 31;
  const int wv   = threadIdx.x >> 5;
#pragma unroll
  for (int sub = 0; sub < 2; ++sub)
#pragma unroll
    for (int r = 0; r < 8; ++r) {
      TILE_MN(ml, nl, sub, r);
      const int m = m0 + ml;
      y[(long)bk * C_ * 64 + (long)m * 64 + nl] = acc[sub][r] + vd_b1[m];
    }
}

// ---- bilinear 8->64 + GELU + attn-weighted sum over k ---------------------
__global__ void __launch_bounds__(256) k_upsample(
    const float* __restrict__ y, const float* __restrict__ attn,
    float* __restrict__ g)
{
  const int c = blockIdx.y, b = blockIdx.z;
  const int pix = blockIdx.x * 256 + threadIdx.x;
  const int h = pix >> 6, w = pix & 63;
  const float sy = (h + 0.5f) * 0.125f - 0.5f;
  const float sx = (w + 0.5f) * 0.125f - 0.5f;
  const int y0i = (int)floorf(sy); const float fy = sy - y0i;
  const int x0i = (int)floorf(sx); const float fx = sx - x0i;
  const int y0 = max(y0i, 0), y1 = min(y0i + 1, VR_ - 1);
  const int x0 = max(x0i, 0), x1 = min(x0i + 1, VR_ - 1);
  float outv = 0.0f;
#pragma unroll
  for (int kk = 0; kk < 2; ++kk) {
    const float* yp = y + ((long)(b * 2 + kk) * C_ + c) * 64;
    const float v00 = yp[y0 * 8 + x0], v01 = yp[y0 * 8 + x1];
    const float v10 = yp[y1 * 8 + x0], v11 = yp[y1 * 8 + x1];
    const float va = v00 + fx * (v01 - v00);
    const float vb = v10 + fx * (v11 - v10);
    outv += attn[b * 2 + kk] * gelu_f(va + fy * (vb - va));
  }
  g[((long)b * C_ + c) * HW_ + pix] = outv;
}

// ---- plain depthwise 3x3, zero padding ------------------------------------
__global__ void __launch_bounds__(256) k_dwconv(
    const float* __restrict__ in, const float* __restrict__ wdw,
    const float* __restrict__ bdw, float* __restrict__ out)
{
  const int c = blockIdx.y, b = blockIdx.z;
  const int pix = blockIdx.x * 256 + threadIdx.x;
  const int h = pix >> 6, w = pix & 63;
  const float* ip = in + ((long)b * C_ + c) * HW_;
  const float* wp = wdw + c * 9;
  float s = bdw[c];
#pragma unroll
  for (int dy = -1; dy <= 1; ++dy)
#pragma unroll
    for (int dx = -1; dx <= 1; ++dx) {
      const int hh = h + dy, ww = w + dx;
      const float v = (hh >= 0 && hh < 64 && ww >= 0 && ww < 64)
                          ? ip[hh * 64 + ww] : 0.0f;
      s += v * wp[(dy + 1) * 3 + (dx + 1)];
    }
  out[((long)b * C_ + c) * HW_ + pix] = s;
}

// ---- fused conv1x1 over cat = [x, wref]  (K = 1280 split) -----------------
__global__ void __launch_bounds__(256) k_fuse1(
    const float* __restrict__ x, const float* __restrict__ wref,
    const float* __restrict__ fn_w1, const float* __restrict__ fn_b1,
    float* __restrict__ f1)
{
  __shared__ __align__(16) half_t As[64 * LDA_];
  __shared__ __align__(16) half_t Bs[64 * LDB_];
  const int p0 = blockIdx.x * 64, m0 = blockIdx.y * 64, b = blockIdx.z;
  v8f acc[2] = {};
  gemm_acc(As, Bs, fn_w1, 2 * C_, 0,  x    + (long)b * C_ * HW_, HW_, m0, C_, p0, C_, acc);
  gemm_acc(As, Bs, fn_w1, 2 * C_, C_, wref + (long)b * C_ * HW_, HW_, m0, C_, p0, C_, acc);
  const int lane = threadIdx.x & 31;
  const int wv   = threadIdx.x >> 5;
#pragma unroll
  for (int sub = 0; sub < 2; ++sub)
#pragma unroll
    for (int r = 0; r < 8; ++r) {
      TILE_MN(ml, nl, sub, r);
      const int m = m0 + ml, p = p0 + nl;
      f1[((long)b * C_ + m) * HW_ + p] = acc[sub][r] + fn_b1[m];
    }
}

// ---- GroupNorm statistics (per batch, per group of 20 channels) -----------
__global__ void __launch_bounds__(256) k_gnstats(
    const float* __restrict__ f1, float* __restrict__ stats)
{
  const int g = blockIdx.x, b = blockIdx.y;
  const float* p = f1 + ((long)b * C_ + g * CPG_) * HW_;
  const long N = (long)CPG_ * HW_;
  float s = 0.0f, s2 = 0.0f;
  for (long i = threadIdx.x; i < N; i += 256) {
    const float v = p[i]; s += v; s2 += v * v;
  }
  __shared__ float sh[256], sh2[256];
  sh[threadIdx.x] = s; sh2[threadIdx.x] = s2;
  __syncthreads();
  for (int st = 128; st > 0; st >>= 1) {
    if (threadIdx.x < st) {
      sh[threadIdx.x]  += sh[threadIdx.x + st];
      sh2[threadIdx.x] += sh2[threadIdx.x + st];
    }
    __syncthreads();
  }
  if (threadIdx.x == 0) {
    const float mu  = sh[0] / (float)N;
    const float var = sh2[0] / (float)N - mu * mu;
    stats[(b * GROUPS_ + g) * 2 + 0] = mu;
    stats[(b * GROUPS_ + g) * 2 + 1] = rsqrtf(var + 1e-5f);
  }
}

// ---- GN + GELU applied on the fly, then depthwise 3x3 ---------------------
__global__ void __launch_bounds__(256) k_gn_dwconv(
    const float* __restrict__ f1, const float* __restrict__ stats,
    const float* __restrict__ gamma, const float* __restrict__ beta,
    const float* __restrict__ wdw, const float* __restrict__ bdw,
    float* __restrict__ out)
{
  const int c = blockIdx.y, b = blockIdx.z;
  const int pix = blockIdx.x * 256 + threadIdx.x;
  const int h = pix >> 6, w = pix & 63;
  const int g = c / CPG_;
  const float mu = stats[(b * GROUPS_ + g) * 2 + 0];
  const float rs = stats[(b * GROUPS_ + g) * 2 + 1];
  const float ga = gamma[c], be = beta[c];
  const float* ip = f1 + ((long)b * C_ + c) * HW_;
  const float* wp = wdw + c * 9;
  float s = bdw[c];
#pragma unroll
  for (int dy = -1; dy <= 1; ++dy)
#pragma unroll
    for (int dx = -1; dx <= 1; ++dx) {
      const int hh = h + dy, ww = w + dx;
      float t = 0.0f;
      if (hh >= 0 && hh < 64 && ww >= 0 && ww < 64) {
        const float v = ip[hh * 64 + ww];
        t = gelu_f((v - mu) * rs * ga + be);
      }
      s += t * wp[(dy + 1) * 3 + (dx + 1)];
    }
  out[((long)b * C_ + c) * HW_ + pix] = s;
}

// ---- final: f2 = conv1x1(fdw, fn_w2); gate = sigmoid(conv1x1(cat, fg_w));
//      out = gate*f2 + (1-gate)*x  ------------------------------------------
__global__ void __launch_bounds__(256) k_final(
    const float* __restrict__ x, const float* __restrict__ wref,
    const float* __restrict__ fdw,
    const float* __restrict__ fn_w2, const float* __restrict__ fn_b2,
    const float* __restrict__ fg_w,  const float* __restrict__ fg_b,
    float* __restrict__ out)
{
  __shared__ __align__(16) half_t As[64 * LDA_];
  __shared__ __align__(16) half_t Bs[64 * LDB_];
  const int p0 = blockIdx.x * 64, m0 = blockIdx.y * 64, b = blockIdx.z;
  v8f accF[2] = {};
  v8f accG[2] = {};
  gemm_acc(As, Bs, fn_w2, C_,     0,  fdw  + (long)b * C_ * HW_, HW_, m0, C_, p0, C_, accF);
  gemm_acc(As, Bs, fg_w,  2 * C_, 0,  x    + (long)b * C_ * HW_, HW_, m0, C_, p0, C_, accG);
  gemm_acc(As, Bs, fg_w,  2 * C_, C_, wref + (long)b * C_ * HW_, HW_, m0, C_, p0, C_, accG);
  const int lane = threadIdx.x & 31;
  const int wv   = threadIdx.x >> 5;
#pragma unroll
  for (int sub = 0; sub < 2; ++sub)
#pragma unroll
    for (int r = 0; r < 8; ++r) {
      TILE_MN(ml, nl, sub, r);
      const int m = m0 + ml, p = p0 + nl;
      const float f2   = accF[sub][r] + fn_b2[m];
      const float gate = sigmoid_f(accG[sub][r] + fg_b[m]);
      const float xv   = x[((long)b * C_ + m) * HW_ + p];
      out[((long)b * C_ + m) * HW_ + p] = gate * f2 + (1.0f - gate) * xv;
    }
}

// ---------------------------------------------------------------------------
extern "C" void kernel_launch(void* const* d_in, const int* in_sizes, int n_in,
                              void* d_out, int out_size, void* d_ws, size_t ws_size,
                              hipStream_t stream)
{
  (void)in_sizes; (void)n_in; (void)out_size; (void)ws_size;
  const float* x          = (const float*)d_in[0];
  const float* qp_w       = (const float*)d_in[1];
  const float* qp_b       = (const float*)d_in[2];
  const float* mem_keys   = (const float*)d_in[3];
  const float* mem_values = (const float*)d_in[4];
  const float* vd_w1      = (const float*)d_in[5];
  const float* vd_b1      = (const float*)d_in[6];
  const float* vd_wdw     = (const float*)d_in[7];
  const float* vd_bdw     = (const float*)d_in[8];
  const float* fn_w1      = (const float*)d_in[9];
  const float* fn_b1      = (const float*)d_in[10];
  const float* fn_gamma   = (const float*)d_in[11];
  const float* fn_beta    = (const float*)d_in[12];
  const float* fn_wdw     = (const float*)d_in[13];
  const float* fn_bdw     = (const float*)d_in[14];
  const float* fn_w2      = (const float*)d_in[15];
  const float* fn_b2      = (const float*)d_in[16];
  const float* fg_w       = (const float*)d_in[17];
  const float* fg_b       = (const float*)d_in[18];
  float* out = (float*)d_out;

  // workspace layout (floats)
  float* ws_f  = (float*)d_ws;
  float* qsum  = ws_f;                         // 8*160
  float* attn  = ws_f + 1280;                  // 16
  int*   idx   = (int*)(ws_f + 1296);          // 16
  float* stats = ws_f + 1312;                  // 8*32*2
  float* ybuf  = ws_f + 2048;                  // 16*640*64
  float* bufA  = ws_f + 657408;                // g, then reused as f1
  float* bufB  = bufA + (size_t)B_ * C_ * HW_; // wref
  float* bufC  = bufB + (size_t)B_ * C_ * HW_; // fdw

  k_zero<<<dim3(5), dim3(256), 0, stream>>>(qsum, B_ * KD_);
  k_qproj<<<dim3(HW_ / 64, 3, B_), dim3(256), 0, stream>>>(x, qp_w, qp_b, qsum);
  k_attn<<<dim3(1), dim3(32), 0, stream>>>(qsum, mem_keys, attn, idx);
  k_refproj<<<dim3(1, C_ / 64, B_ * 2), dim3(256), 0, stream>>>(
      mem_values, vd_w1, vd_b1, idx, ybuf);
  k_upsample<<<dim3(HW_ / 256, C_, B_), dim3(256), 0, stream>>>(ybuf, attn, bufA);
  k_dwconv<<<dim3(HW_ / 256, C_, B_), dim3(256), 0, stream>>>(
      bufA, vd_wdw, vd_bdw, bufB);
  k_fuse1<<<dim3(HW_ / 64, C_ / 64, B_), dim3(256), 0, stream>>>(
      x, bufB, fn_w1, fn_b1, bufA);
  k_gnstats<<<dim3(GROUPS_, B_), dim3(256), 0, stream>>>(bufA, stats);
  k_gn_dwconv<<<dim3(HW_ / 256, C_, B_), dim3(256), 0, stream>>>(
      bufA, stats, fn_gamma, fn_beta, fn_wdw, fn_bdw, bufC);
  k_final<<<dim3(HW_ / 64, C_ / 64, B_), dim3(256), 0, stream>>>(
      x, bufB, bufC, fn_w2, fn_b2, fg_w, fg_b, out);
}
